// BaseConvLSTMRegressor_89678917140569
// MI455X (gfx1250) — compile-verified, matrix operand order checked by reading, not run on hardware
//
#include <hip/hip_runtime.h>

#define B_  16
#define T_  24
#define H_  96
#define W_  96
#define CIN 16
#define F_  16
#define NG  64   // 4*FILTERS gate channels

typedef __bf16 bf16t;
typedef __attribute__((ext_vector_type(16))) bf16t v16bf;
typedef __attribute__((ext_vector_type(8)))  float v8f;

__device__ __forceinline__ unsigned pkbf(float lo, float hi) {
  bf16t l = (bf16t)lo, h = (bf16t)hi;
  unsigned short lu = __builtin_bit_cast(unsigned short, l);
  unsigned short hu = __builtin_bit_cast(unsigned short, h);
  return (unsigned)lu | ((unsigned)hu << 16);
}
__device__ __forceinline__ float sigmoidf(float x) {
  return 1.0f / (1.0f + __expf(-x));
}

// ---------------------------------------------------------------------------
// Pack kernel(3,3,16,64) + recurrent_kernel(3,3,16,64) into the WMMA B-matrix
// lane layout (N-major, same striping as the 16-bit A layout), bf16 pairs.
// dword index = ((kc*4 + nt)*32 + lane)*8 + v
// ---------------------------------------------------------------------------
__global__ void pack_weights(const float* __restrict__ kern,
                             const float* __restrict__ rkern,
                             unsigned* __restrict__ wpack) {
  int idx = blockIdx.x * blockDim.x + threadIdx.x;
  if (idx >= 9 * 4 * 32 * 8) return;
  int v    = idx & 7;
  int lane = (idx >> 3) & 31;
  int nt   = (idx >> 8) & 3;
  int kc   = idx >> 10;                  // tap index 0..8
  int half = lane >> 4;
  int n    = nt * 16 + (lane & 15);
  int k    = (v < 4) ? (v * 2 + half * 8) : (16 + (v - 4) * 2 + half * 8);
  float w0, w1;
  if (k < 16) {                          // channels 0..15 -> kernel (x path)
    w0 = kern[(kc * 16 + k) * NG + n];
    w1 = kern[(kc * 16 + k + 1) * NG + n];
  } else {                               // channels 16..31 -> recurrent kernel
    w0 = rkern[(kc * 16 + (k - 16)) * NG + n];
    w1 = rkern[(kc * 16 + (k - 15)) * NG + n];
  }
  wpack[idx] = pkbf(w0, w1);
}

__global__ void zero_f32(float* __restrict__ p, int n) {
  int i = blockIdx.x * blockDim.x + threadIdx.x;
  if (i < n) p[i] = 0.0f;
}

// ---------------------------------------------------------------------------
// One ConvLSTM step: z = conv(x_t,K) + conv(h_prev,RK) + bias as a fused
// implicit GEMM (M=pixels, N=64 gates, K=9 taps * 32 ch) via bf16 WMMA.
// Block = 128 threads = 4 waves; each wave owns a 16-pixel x 64-gate tile.
// Input halo (3 rows x 18 cols x 32ch) staged once per wave into LDS as
// packed bf16, so the WMMA inner loop is pure ds_load_b128 + v_wmma.
// ---------------------------------------------------------------------------
__global__ __launch_bounds__(128)
void convlstm_step(const float* __restrict__ x, int t,
                   const unsigned* __restrict__ wpack,
                   const float* __restrict__ bias,
                   const float* __restrict__ hprev,
                   const float* __restrict__ cprev,
                   float* __restrict__ hnew,
                   float* __restrict__ cnew) {
  __shared__ uint4 wlds[2304];              // 36 KB packed weights
  __shared__ unsigned itile[4 * 3 * 18 * 16]; // 13.5 KB input halo tiles
  __shared__ float zbuf[4][16][NG];         // 16 KB z tiles
  __shared__ float blds[NG];                // bias

  const int tid = threadIdx.x;
  // Stage weights + bias once per block.
  const uint4* wsrc = (const uint4*)wpack;
  for (int i = tid; i < 2304; i += 128) wlds[i] = wsrc[i];
  if (tid < NG) blds[tid] = bias[tid];

  const int wave = tid >> 5;
  const int lane = tid & 31;
  const int half = lane >> 4;
  const int mrow = lane & 15;

  const int tile = blockIdx.x * 4 + wave;     // 0..9215
  const int wt   = tile % (W_ / 16);
  const int rest = tile / (W_ / 16);
  const int hh   = rest % H_;
  const int b    = rest / H_;
  const int w0   = wt * 16;

  const float* xb = x + ((long)(b * T_ + t) * H_ * W_) * CIN;
  const float* hb = hprev + ((long)b * H_ * W_) * CIN;

  // ---- Stage this wave's 3x18 halo (32 bf16 channels per site) into LDS ----
  for (int s = lane; s < 54; s += 32) {
    const int r    = s / 18;                  // halo row 0..2
    const int ccol = s % 18;                  // halo col 0..17
    const int sy   = hh + r - 1;
    const int sx   = w0 + ccol - 1;
    const bool inb = ((unsigned)sy < (unsigned)H_) & ((unsigned)sx < (unsigned)W_);
    // Clamp so loads are always in-bounds; zero the result via selects.
    const int syc = min(max(sy, 0), H_ - 1);
    const int sxc = min(max(sx, 0), W_ - 1);
    const float4* xp = (const float4*)(xb + ((long)syc * W_ + sxc) * CIN);
    const float4* hp = (const float4*)(hb + ((long)syc * W_ + sxc) * CIN);
    float4 xv0 = xp[0], xv1 = xp[1], xv2 = xp[2], xv3 = xp[3];
    float4 hv0 = hp[0], hv1 = hp[1], hv2 = hp[2], hv3 = hp[3];
    unsigned d[16];
    d[0]  = pkbf(xv0.x, xv0.y); d[1]  = pkbf(xv0.z, xv0.w);
    d[2]  = pkbf(xv1.x, xv1.y); d[3]  = pkbf(xv1.z, xv1.w);
    d[4]  = pkbf(xv2.x, xv2.y); d[5]  = pkbf(xv2.z, xv2.w);
    d[6]  = pkbf(xv3.x, xv3.y); d[7]  = pkbf(xv3.z, xv3.w);
    d[8]  = pkbf(hv0.x, hv0.y); d[9]  = pkbf(hv0.z, hv0.w);
    d[10] = pkbf(hv1.x, hv1.y); d[11] = pkbf(hv1.z, hv1.w);
    d[12] = pkbf(hv2.x, hv2.y); d[13] = pkbf(hv2.z, hv2.w);
    d[14] = pkbf(hv3.x, hv3.y); d[15] = pkbf(hv3.z, hv3.w);
    const unsigned zmask = inb ? 0xFFFFFFFFu : 0u;
    uint4* dst = (uint4*)&itile[((wave * 3 + r) * 18 + ccol) * 16];
    dst[0] = make_uint4(d[0] & zmask,  d[1] & zmask,  d[2] & zmask,  d[3] & zmask);
    dst[1] = make_uint4(d[4] & zmask,  d[5] & zmask,  d[6] & zmask,  d[7] & zmask);
    dst[2] = make_uint4(d[8] & zmask,  d[9] & zmask,  d[10] & zmask, d[11] & zmask);
    dst[3] = make_uint4(d[12] & zmask, d[13] & zmask, d[14] & zmask, d[15] & zmask);
  }
  __syncthreads();

  // ---- WMMA main loop: pure LDS b128 loads + v_wmma ----
  v8f acc[4];
  acc[0] = v8f{}; acc[1] = v8f{}; acc[2] = v8f{}; acc[3] = v8f{};

  // A-fragment base for this lane: site (row 0, col mrow), dword half*4.
  const uint4* abase = (const uint4*)&itile[((wave * 3 + 0) * 18 + mrow) * 16 + half * 4];

  #pragma unroll
  for (int kc = 0; kc < 9; ++kc) {
    const int dy = kc / 3;                   // halo row 0..2
    const int dx = kc % 3;                   // halo col offset 0..2
    const int o  = (dy * 18 + dx) * 4;       // uint4 units (16 dwords/site)
    union { v16bf v; uint4 q[2]; } a;
    a.q[0] = abase[o];
    a.q[1] = abase[o + 2];
    #pragma unroll
    for (int nt = 0; nt < 4; ++nt) {
      union { v16bf v; uint4 q[2]; } bf;
      const int base = ((kc * 4 + nt) * 32 + lane) * 2;
      bf.q[0] = wlds[base];
      bf.q[1] = wlds[base + 1];
      acc[nt] = __builtin_amdgcn_wmma_f32_16x16x32_bf16(
          false, a.v, false, bf.v, (short)0, acc[nt], false, false);
    }
  }

  // Spill accumulators: C/D layout -> lane holds column n, rows r + 8*half.
  #pragma unroll
  for (int nt = 0; nt < 4; ++nt)
    #pragma unroll
    for (int r = 0; r < 8; ++r)
      zbuf[wave][r + 8 * half][nt * 16 + mrow] = acc[nt][r];
  __syncthreads();

  // Gate math: 256 (pixel,filter) outputs per wave, 8 per lane.
  const long pixbase = ((long)b * H_ + hh) * W_ + w0;
  #pragma unroll
  for (int j = 0; j < 8; ++j) {
    const int idx = lane * 8 + j;
    const int p = idx >> 4;
    const int f = idx & 15;
    const float zi = zbuf[wave][p][f]        + blds[f];
    const float zf = zbuf[wave][p][16 + f]   + blds[16 + f];
    const float zc = zbuf[wave][p][32 + f]   + blds[32 + f];
    const float zo = zbuf[wave][p][48 + f]   + blds[48 + f];
    const float ig = sigmoidf(zi);
    const float fg = sigmoidf(zf);
    const float og = sigmoidf(zo);
    const long off = (pixbase + p) * F_ + f;
    const float cp = cprev[off];
    const float cn = fg * cp + ig * fmaxf(zc, 0.0f);
    cnew[off] = cn;
    hnew[off] = og * fmaxf(cn, 0.0f);
  }
}

// ---------------------------------------------------------------------------
// Dense(1): out[b,h,w] = h_last[b,h,w,:] . dense_w + dense_b
// ---------------------------------------------------------------------------
__global__ void dense_out(const float* __restrict__ h,
                          const float* __restrict__ dw,
                          const float* __restrict__ db,
                          float* __restrict__ out, int n) {
  int i = blockIdx.x * blockDim.x + threadIdx.x;
  if (i >= n) return;
  const float* hp = h + (long)i * F_;
  float s = db[0];
  #pragma unroll
  for (int c = 0; c < F_; ++c) s += hp[c] * dw[c];
  out[i] = s;
}

extern "C" void kernel_launch(void* const* d_in, const int* in_sizes, int n_in,
                              void* d_out, int out_size, void* d_ws, size_t ws_size,
                              hipStream_t stream) {
  const float* inputs = (const float*)d_in[0];   // (B,T,H,W,16)
  const float* kern   = (const float*)d_in[1];   // (3,3,16,64)
  const float* rkern  = (const float*)d_in[2];   // (3,3,16,64)
  const float* bias   = (const float*)d_in[3];   // (64,)
  const float* dw     = (const float*)d_in[4];   // (16,1)
  const float* db     = (const float*)d_in[5];   // (1,)
  float* out = (float*)d_out;                    // (B,96,96,1)

  char* ws = (char*)d_ws;
  unsigned* wpack = (unsigned*)ws;               // 9216 dwords
  size_t off = (9216u * sizeof(unsigned) + 255u) & ~(size_t)255u;
  const size_t HC = (size_t)B_ * H_ * W_ * F_;   // elements per state buffer
  float* h0 = (float*)(ws + off); off += HC * sizeof(float);
  float* c0 = (float*)(ws + off); off += HC * sizeof(float);
  float* h1 = (float*)(ws + off); off += HC * sizeof(float);
  float* c1 = (float*)(ws + off); off += HC * sizeof(float);

  pack_weights<<<36, 256, 0, stream>>>(kern, rkern, wpack);
  const int zn = (int)(2 * HC);                  // h0 and c0 are contiguous
  zero_f32<<<(zn + 255) / 256, 256, 0, stream>>>(h0, zn);

  float* hc = h0; float* cc = c0; float* hn = h1; float* cn = c1;
  for (int t = 0; t < T_; ++t) {
    convlstm_step<<<2304, 128, 0, stream>>>(inputs, t, wpack, bias, hc, cc, hn, cn);
    float* th = hc; hc = hn; hn = th;
    float* tc = cc; cc = cn; cn = tc;
  }

  const int npix = B_ * H_ * W_;
  dense_out<<<(npix + 255) / 256, 256, 0, stream>>>(hc, dw, db, out, npix);
}
